// AdaptiveAttention_40389872451959
// MI455X (gfx1250) — matrix operand order repaired and checked by reading, not compiled, not run on hardware
//
#include <hip/hip_runtime.h>

// ---------------- problem constants ----------------
#define B_DIM 4096
#define H_DIM 1024
#define P_DIM 49
#define PPAD  64          // pad p/q to 64 -> 4x4 WMMA tiles, no edge handling
#define HPAD  1032        // pad K rows to 1032 elems to stagger LDS banks

typedef __attribute__((ext_vector_type(16))) __bf16     v16bf;
typedef __attribute__((ext_vector_type(8)))  float      v8f;
typedef __attribute__((ext_vector_type(4)))  unsigned   u32x4;

// ---------------- LDS layout (bytes) ----------------
#define OFF_VT   0                            // V[b]^T as bf16 [p=0..63][h=0..1031]
#define SZ_BIG   (PPAD * HPAD * 2)            // 132096
#define OFF_WV   (OFF_VT + SZ_BIG)            // Wv as bf16 [q][h]
#define OFF_HT   (OFF_WV + SZ_BIG)            // ht[b] f32
#define OFF_ST   (OFF_HT + H_DIM * 4)         // st[b] f32
#define OFF_HP   (OFF_ST + H_DIM * 4)         // hp[64] f32
#define OFF_SP   (OFF_HP + 64 * 4)            // sp[64] f32
#define OFF_Z    (OFF_SP + 64 * 4)            // z[64]  f32
#define OFF_AL   (OFF_Z  + 64 * 4)            // alpha[64] f32
#define OFF_SC   (OFF_AL + 64 * 4)            // scalars
#define SMEM_BYTES (OFF_SC + 64)              // 273472 B < 320 KB/WGP

__device__ __forceinline__ unsigned short f32_bf16(float f) {
    unsigned u = __float_as_uint(f);
    u += 0x7FFFu + ((u >> 16) & 1u);          // round-to-nearest-even
    return (unsigned short)(u >> 16);
}
__device__ __forceinline__ float bf16_f32(unsigned short s) {
    return __uint_as_float(((unsigned)s) << 16);
}
__device__ __forceinline__ float wave_sum(float v) {
    #pragma unroll
    for (int m = 16; m; m >>= 1) v += __shfl_xor(v, m, 32);
    return v;
}
__device__ __forceinline__ float wave_max(float v) {
    #pragma unroll
    for (int m = 16; m; m >>= 1) v = fmaxf(v, __shfl_xor(v, m, 32));
    return v;
}
__device__ __forceinline__ float half_sum(float v) {   // reduce within groups of 16 lanes
    #pragma unroll
    for (int m = 8; m; m >>= 1) v += __shfl_xor(v, m, 32);
    return v;
}

extern "C" __global__ __launch_bounds__(256, 1)
void adaptive_attn_kernel(const float* __restrict__ V,    const float* __restrict__ ht,
                          const float* __restrict__ st,   const float* __restrict__ Wv_w,
                          const float* __restrict__ Wv_b, const float* __restrict__ Ws_w,
                          const float* __restrict__ Ws_b, const float* __restrict__ Wg_w,
                          const float* __restrict__ wh_w,
                          float* __restrict__ out_c,     float* __restrict__ out_ctx,
                          float* __restrict__ out_alpha, float* __restrict__ out_beta)
{
    extern __shared__ char smem[];
    unsigned short* vt = (unsigned short*)(smem + OFF_VT);
    unsigned short* wv = (unsigned short*)(smem + OFF_WV);
    float* hts = (float*)(smem + OFF_HT);
    float* sts = (float*)(smem + OFF_ST);
    float* hp  = (float*)(smem + OFF_HP);
    float* sp  = (float*)(smem + OFF_SP);
    float* zz  = (float*)(smem + OFF_Z);
    float* al  = (float*)(smem + OFF_AL);
    float* sc  = (float*)(smem + OFF_SC);

    const int t    = threadIdx.x;
    const int b    = blockIdx.x;
    const int wave = t >> 5;
    const int lane = t & 31;

    // ---------- Phase 0: single-pass HBM read of V[b]; stage everything in LDS ----------
    {
        const float* Vb = V + (size_t)b * (H_DIM * P_DIM);
        for (int f = t; f < H_DIM * P_DIM; f += 256) {          // coalesced global read
            int h = f / P_DIM, p = f - h * P_DIM;
            vt[p * HPAD + h] = f32_bf16(Vb[f]);                 // transposed bf16 store
        }
        for (int f = t; f < P_DIM * H_DIM; f += 256) {          // Wv_w is [q][h] already
            int q = f >> 10, h = f & (H_DIM - 1);
            wv[q * HPAD + h] = f32_bf16(Wv_w[f]);
        }
        for (int f = t; f < (PPAD - P_DIM) * HPAD; f += 256) {  // zero pad rows 49..63
            int r = P_DIM + f / HPAD, c = f % HPAD;
            vt[r * HPAD + c] = 0;
            wv[r * HPAD + c] = 0;
        }
        for (int h = t; h < H_DIM; h += 256) {
            hts[h] = ht[(size_t)b * H_DIM + h];
            sts[h] = st[(size_t)b * H_DIM + h];
        }
        if (t < 64) { hp[t] = 0.f; sp[t] = 0.f; zz[t] = 0.f; }
    }
    __syncthreads();

    // ---------- Phase 1: hp[q] = ht·Wg[q],  sp[q] = st·Ws[q] + Ws_b[q] ----------
    for (int q = wave; q < P_DIM; q += 8) {                     // wave-parallel dots
        const float* wg = Wg_w + q * H_DIM;
        const float* ws = Ws_w + q * H_DIM;
        float ah = 0.f, as = 0.f;
        #pragma unroll 8
        for (int j = 0; j < H_DIM / 32; ++j) {
            int h = lane + 32 * j;                              // coalesced (L2-hot weights)
            ah = fmaf(hts[h], wg[h], ah);
            as = fmaf(sts[h], ws[h], as);
        }
        ah = wave_sum(ah);
        as = wave_sum(as);
        if (lane == 0) { hp[q] = ah; sp[q] = as + Ws_b[q]; }
    }
    __syncthreads();

    // ---------- Phase 2: img_proj = V[b]^T · Wv^T  via v_wmma_f32_16x16x32_bf16 ----------
    {
        const int pt  = wave & 3;            // p tile  (rows of C)
        const int qt0 = (wave >> 2) * 2;     // first of 2 q tiles owned by this wave
        const int hi  = lane >> 4;           // half-wave selector
        const int nl  = lane & 15;

        v8f acc0 = {}; v8f acc1 = {};
        // A (16x32 bf16): lane row p = pt*16+nl; K 0..15 (lo half) / 16..31 (hi half)
        const char* abase  = (const char*)(vt + (pt * 16 + nl) * HPAD + hi * 16);
        // B (32x16 bf16): lane col q = qt*16+nl; K 0..15 (lo half) / 16..31 (hi half)
        const char* bbase0 = (const char*)(wv + ((qt0    ) * 16 + nl) * HPAD + hi * 16);
        const char* bbase1 = (const char*)(wv + ((qt0 + 1) * 16 + nl) * HPAD + hi * 16);

        for (int kb = 0; kb < H_DIM; kb += 32) {
            union { v16bf bf; u32x4 u[2]; } A, B0, B1;
            A.u[0]  = *(const u32x4*)(abase  + kb * 2);         // K = kb+hi*16 + 0..7
            A.u[1]  = *(const u32x4*)(abase  + kb * 2 + 16);    // K = kb+hi*16 + 8..15
            B0.u[0] = *(const u32x4*)(bbase0 + kb * 2);
            B0.u[1] = *(const u32x4*)(bbase0 + kb * 2 + 16);
            B1.u[0] = *(const u32x4*)(bbase1 + kb * 2);
            B1.u[1] = *(const u32x4*)(bbase1 + kb * 2 + 16);
            acc0 = __builtin_amdgcn_wmma_f32_16x16x32_bf16(false, A.bf, false, B0.bf,
                                                           (short)0, acc0, false, false);
            acc1 = __builtin_amdgcn_wmma_f32_16x16x32_bf16(false, A.bf, false, B1.bf,
                                                           (short)0, acc1, false, false);
        }

        // epilogue: z[p] += sum_q tanh(img + Wv_b[q] + hp[p]) * wh[q]
        #pragma unroll
        for (int i = 0; i < 2; ++i) {
            v8f acc = i ? acc1 : acc0;
            int q = (qt0 + i) * 16 + nl;
            float whq = (q < P_DIM) ? wh_w[q] : 0.f;            // zero kills padded cols
            float bq  = (q < P_DIM) ? Wv_b[q] : 0.f;
            #pragma unroll
            for (int r = 0; r < 8; ++r) {
                int p = pt * 16 + hi * 8 + r;                   // C layout: VGPR r -> M=r(+8)
                float term = tanhf(acc[r] + bq + hp[p]) * whq;
                term = half_sum(term);                          // reduce 16 q-lanes
                if (nl == 0) atomicAdd(&zz[p], term);           // ds_add_f32
            }
        }
    }
    __syncthreads();

    // ---------- Phase 3: softmaxes (wave 0) ----------
    if (wave == 0) {
        int p0 = lane, p1 = lane + 32;
        float z0 = zz[p0];                                      // p0 <= 31 < 49 always valid
        float z1 = (p1 < P_DIM) ? zz[p1] : -3.0e38f;
        float m1 = wave_max(fmaxf(z0, z1));
        float e0 = __expf(z0 - m1);
        float e1 = (p1 < P_DIM) ? __expf(z1 - m1) : 0.f;
        float s1 = wave_sum(e0 + e1);
        // attn_vs = sum_q tanh(sp[q] + hp[q]) * wh[q]
        float a0 = tanhf(sp[p0] + hp[p0]) * wh_w[p0];
        float a1 = (p1 < P_DIM) ? tanhf(sp[p1] + hp[p1]) * wh_w[p1] : 0.f;
        float av = wave_sum(a0 + a1);
        // concat softmax -> beta
        float m2    = fmaxf(m1, av);
        float denom = s1 * __expf(m1 - m2) + __expf(av - m2);
        float beta  = __expf(av - m2) / denom;
        float inv_s1 = 1.f / s1;
        al[p0] = e0 * inv_s1;
        out_alpha[(size_t)b * P_DIM + p0] = e0 * inv_s1;
        if (p1 < P_DIM) {
            al[p1] = e1 * inv_s1;
            out_alpha[(size_t)b * P_DIM + p1] = e1 * inv_s1;
        }
        if (lane == 0) { sc[0] = beta; out_beta[b] = beta; }
    }
    __syncthreads();

    // ---------- Phase 4: context_t from LDS-resident V; c_t_hat; coalesced stores ----------
    {
        float beta = sc[0];
        float ob   = 1.f - beta;
        #pragma unroll
        for (int j = 0; j < H_DIM / 256; ++j) {
            int h = t + j * 256;
            float ctx = 0.f;
            #pragma unroll 7
            for (int p = 0; p < P_DIM; ++p)
                ctx = fmaf(bf16_f32(vt[p * HPAD + h]), al[p], ctx);
            size_t o = (size_t)b * H_DIM + h;
            out_ctx[o] = ctx;
            out_c[o]   = beta * sts[h] + ob * ctx;
        }
    }
}

extern "C" void kernel_launch(void* const* d_in, const int* in_sizes, int n_in,
                              void* d_out, int out_size, void* d_ws, size_t ws_size,
                              hipStream_t stream) {
    const float* V    = (const float*)d_in[0];
    const float* ht   = (const float*)d_in[1];
    const float* st   = (const float*)d_in[2];
    const float* Wv_w = (const float*)d_in[3];
    const float* Wv_b = (const float*)d_in[4];
    const float* Ws_w = (const float*)d_in[5];
    const float* Ws_b = (const float*)d_in[6];
    const float* Wg_w = (const float*)d_in[7];
    const float* wh_w = (const float*)d_in[8];

    float* out       = (float*)d_out;
    float* out_c     = out;                                   // (B, H)
    float* out_ctx   = out + (size_t)B_DIM * H_DIM;           // (B, H)
    float* out_alpha = out + 2 * (size_t)B_DIM * H_DIM;       // (B, P)
    float* out_beta  = out_alpha + (size_t)B_DIM * P_DIM;     // (B, 1)

    (void)hipFuncSetAttribute((const void*)adaptive_attn_kernel,
                              hipFuncAttributeMaxDynamicSharedMemorySize, SMEM_BYTES);

    adaptive_attn_kernel<<<dim3(B_DIM), dim3(256), SMEM_BYTES, stream>>>(
        V, ht, st, Wv_w, Wv_b, Ws_w, Ws_b, Wg_w, wh_w,
        out_c, out_ctx, out_alpha, out_beta);
}
